// VQ_10943576670576
// MI455X (gfx1250) — compile-verified, matrix operand order checked by reading, not run on hardware
//
#include <hip/hip_runtime.h>

#define B_     16384
#define DIN_   1024
#define D_     512
#define K_     8192
#define NSPLIT 4
#define NT     32   // codebook codes staged in LDS per block iteration

typedef __attribute__((ext_vector_type(16))) __bf16 v16bf;
typedef __attribute__((ext_vector_type(8)))  __bf16 bf16x8;
typedef __attribute__((ext_vector_type(4)))  __bf16 bf16x4;
typedef __attribute__((ext_vector_type(8)))  float  v8f;
typedef __attribute__((ext_vector_type(4)))  float  f32x4;
typedef __attribute__((ext_vector_type(4)))  unsigned int u32x4;
typedef __attribute__((ext_vector_type(8)))  int    i32x8;
typedef __attribute__((ext_vector_type(4)))  int    i32x4;

// Load a 16-element bf16 A/B fragment for one lane, per the CDNA5 16-bit
// operand layout: elements 0..7 at byte offset 0, elements 8..15 at +32B
// (K = half*8 + {0..7} and K = 16 + half*8 + {0..7}).
__device__ __forceinline__ v16bf load_frag(const __bf16* __restrict__ p) {
  union { v16bf v; bf16x8 h[2]; } u;
  u.h[0] = *(const bf16x8*)(p);
  u.h[1] = *(const bf16x8*)(p + 16);
  return u.v;
}

// ---- Tensor Data Mover: DMA a 2-D tile (NT rows x 512 cols, bf16, row-major)
// from global memory into LDS. D# packed per CDNA5 ISA section 8.3/8.4.
__device__ __forceinline__ void tdm_load_tile(const __bf16* gsrc, unsigned lds_addr) {
  unsigned long long ga = (unsigned long long)(uintptr_t)gsrc;
  u32x4 g0;
  g0.x = 1u;                                   // count=1, user descriptor
  g0.y = lds_addr;                             // lds_addr (bytes), bits[63:32]
  g0.z = (unsigned)ga;                         // global_addr[31:0]
  g0.w = (unsigned)((ga >> 32) & 0x01FFFFFFu)  // global_addr[56:32]
         | (2u << 30);                         // type = 2 ("image")
  i32x8 g1;
  g1[0] = (int)(1u << 16);                     // data_size = 1 -> 2 bytes/elem
  g1[1] = (int)((unsigned)D_ << 16);           // tensor_dim0[15:0] = 512
  g1[2] = (int)((unsigned)K_ << 16);           // tensor_dim0[31:16]=0 | tensor_dim1[15:0]=8192
  g1[3] = (int)((unsigned)D_ << 16);           // tensor_dim1[31:16]=0 | tile_dim0 = 512
  g1[4] = (int)(unsigned)NT;                   // tile_dim1 = NT, tile_dim2 = 0
  g1[5] = (int)D_;                             // tensor_dim0_stride[31:0] = 512
  g1[6] = 0;                                   // dim0_stride[47:32]=0 | dim1_stride lo = 0
  g1[7] = 0;
  i32x4 gz = {0, 0, 0, 0};                     // groups 2/3 unused (2-D tile)
#if defined(__clang_major__) && __clang_major__ >= 23
  i32x8 gz8 = {0, 0, 0, 0, 0, 0, 0, 0};
  __builtin_amdgcn_tensor_load_to_lds(g0, g1, gz, gz, gz8, 0);
#else
  __builtin_amdgcn_tensor_load_to_lds(g0, g1, gz, gz, 0);
#endif
}

// ---------------- kernel 0a: f32 -> bf16 (x4 vectorized) ----------------
__global__ __launch_bounds__(256) void cvt_bf16_kernel(const float* __restrict__ src,
                                                       __bf16* __restrict__ dst, int n4) {
  int i = blockIdx.x * 256 + threadIdx.x;
  if (i < n4) {
    f32x4 v = ((const f32x4*)src)[i];
    bf16x4 o;
    o.x = (__bf16)v.x; o.y = (__bf16)v.y; o.z = (__bf16)v.z; o.w = (__bf16)v.w;
    ((bf16x4*)dst)[i] = o;
  }
}

// ---------------- kernel 0b: codebook -> bf16 + f32 row norms ----------------
__global__ __launch_bounds__(128) void cb_prep_kernel(const float* __restrict__ cbf,
                                                      __bf16* __restrict__ cbb,
                                                      float* __restrict__ cnorm) {
  __shared__ float red[128];
  const int k = blockIdx.x, t = threadIdx.x;
  f32x4 v = ((const f32x4*)(cbf + (size_t)k * D_))[t];
  bf16x4 o;
  o.x = (__bf16)v.x; o.y = (__bf16)v.y; o.z = (__bf16)v.z; o.w = (__bf16)v.w;
  ((bf16x4*)(cbb + (size_t)k * D_))[t] = o;
  red[t] = v.x * v.x + v.y * v.y + v.z * v.z + v.w * v.w;
  __syncthreads();
#pragma unroll
  for (int s = 64; s > 0; s >>= 1) {
    if (t < s) red[t] += red[t + s];
    __syncthreads();
  }
  if (t == 0) cnorm[k] = red[0];
}

// Store one 16x16 f32 accumulator tile per the CDNA5 C/D layout:
// VGPR j holds (M = j + 8*half, N = lane%16).
__device__ __forceinline__ void store_ctile(float* __restrict__ ze, int row0, int col,
                                            const v8f& c, int hf) {
#pragma unroll
  for (int j = 0; j < 8; ++j)
    ze[(size_t)(row0 + 8 * hf + j) * D_ + col] = c[j];
}

// ---------------- kernel 1: z_e = x @ W^T  (bf16 WMMA, f32 acc) ----------------
// block = 256 thr (8 waves, 4x2), block tile 128x64, wave tile 32x32.
__global__ __launch_bounds__(256) void gemm1_kernel(const __bf16* __restrict__ xb,
                                                    const __bf16* __restrict__ wb,
                                                    float* __restrict__ ze) {
  const int lane = threadIdx.x & 31;
  const int wave = threadIdx.x >> 5;
  const int lm = lane & 15;
  const int hf = lane >> 4;
  const int mBase = blockIdx.x * 128 + (wave >> 1) * 32;
  const int nBase = blockIdx.y * 64 + (wave & 1) * 32;

  v8f c00 = {}, c01 = {}, c10 = {}, c11 = {};
  const __bf16* pa0 = xb + (size_t)(mBase + lm) * DIN_ + hf * 8;
  const __bf16* pa1 = pa0 + (size_t)16 * DIN_;
  const __bf16* pb0 = wb + (size_t)(nBase + lm) * DIN_ + hf * 8;
  const __bf16* pb1 = pb0 + (size_t)16 * DIN_;

#pragma unroll 4
  for (int k = 0; k < DIN_; k += 32) {
    v16bf a0 = load_frag(pa0 + k);
    v16bf a1 = load_frag(pa1 + k);
    v16bf b0 = load_frag(pb0 + k);
    v16bf b1 = load_frag(pb1 + k);
    c00 = __builtin_amdgcn_wmma_f32_16x16x32_bf16(false, a0, false, b0, (short)0, c00, false, false);
    c01 = __builtin_amdgcn_wmma_f32_16x16x32_bf16(false, a0, false, b1, (short)0, c01, false, false);
    c10 = __builtin_amdgcn_wmma_f32_16x16x32_bf16(false, a1, false, b0, (short)0, c10, false, false);
    c11 = __builtin_amdgcn_wmma_f32_16x16x32_bf16(false, a1, false, b1, (short)0, c11, false, false);
  }
  store_ctile(ze, mBase,      nBase + lm,      c00, hf);
  store_ctile(ze, mBase,      nBase + 16 + lm, c01, hf);
  store_ctile(ze, mBase + 16, nBase + lm,      c10, hf);
  store_ctile(ze, mBase + 16, nBase + 16 + lm, c11, hf);
}

// ---------------- kernel 2: dist GEMM + fused argmin ----------------
// Each wave holds 16 rows of z_e in registers (16 v16bf frags). The codebook
// is staged through LDS by the Tensor Data Mover: wave 0 DMAs the next
// NT x 512 bf16 tile (32 KB) into the alternate buffer while all 8 waves
// compute wmma from the current one (ds_load_b128 feeds the fragments), so
// global codebook traffic is shared 8-ways. score = |c|^2 - 2*dot.
// grid = (B/128, NSPLIT); 8 waves/block each own 16 rows.
__global__ __launch_bounds__(256) void gemm2_argmin_kernel(const float* __restrict__ ze,
                                                           const __bf16* __restrict__ cbb,
                                                           const float* __restrict__ cnorm,
                                                           float* __restrict__ bestVal,
                                                           int* __restrict__ bestIdx) {
  __shared__ __bf16 smem[2][NT * D_];  // 2 x 32 KB double buffer

  const int lane = threadIdx.x & 31;
  const int wave = threadIdx.x >> 5;
  const int lm = lane & 15;
  const int hf = lane >> 4;
  const int mBase = blockIdx.x * 128 + wave * 16;
  const int n0 = blockIdx.y * (K_ / NSPLIT);
  const int n1 = n0 + (K_ / NSPLIT);

  const unsigned ldsAddr[2] = {(unsigned)(uintptr_t)(&smem[0][0]),
                               (unsigned)(uintptr_t)(&smem[1][0])};

  // Load this wave's 16x512 A tile (f32 -> bf16 in-register), kept live all loop.
  v16bf a[16];
  const float* zrow = ze + (size_t)(mBase + lm) * D_;
#pragma unroll
  for (int s = 0; s < 16; ++s) {
    const float* p = zrow + s * 32 + hf * 8;
    f32x4 v0 = *(const f32x4*)(p);
    f32x4 v1 = *(const f32x4*)(p + 4);
    f32x4 v2 = *(const f32x4*)(p + 16);
    f32x4 v3 = *(const f32x4*)(p + 20);
    union { v16bf v; __bf16 e[16]; } u;
    u.e[0]  = (__bf16)v0.x; u.e[1]  = (__bf16)v0.y; u.e[2]  = (__bf16)v0.z; u.e[3]  = (__bf16)v0.w;
    u.e[4]  = (__bf16)v1.x; u.e[5]  = (__bf16)v1.y; u.e[6]  = (__bf16)v1.z; u.e[7]  = (__bf16)v1.w;
    u.e[8]  = (__bf16)v2.x; u.e[9]  = (__bf16)v2.y; u.e[10] = (__bf16)v2.z; u.e[11] = (__bf16)v2.w;
    u.e[12] = (__bf16)v3.x; u.e[13] = (__bf16)v3.y; u.e[14] = (__bf16)v3.z; u.e[15] = (__bf16)v3.w;
    a[s] = u.v;
  }

  float best[8];
  int bidx[8];
#pragma unroll
  for (int j = 0; j < 8; ++j) { best[j] = 3.0e38f; bidx[j] = 0x7fffffff; }

  // Prologue: DMA first tile, wait, make visible to all waves.
  if (wave == 0) {
    tdm_load_tile(cbb + (size_t)n0 * D_, ldsAddr[0]);
    __builtin_amdgcn_s_wait_tensorcnt(0);
  }
  __syncthreads();

  int cur = 0;
  for (int n = n0; n < n1; n += NT) {
    // Prefetch next tile into the alternate buffer (overlaps compute).
    if (wave == 0 && (n + NT) < n1)
      tdm_load_tile(cbb + (size_t)(n + NT) * D_, ldsAddr[cur ^ 1]);

    const __bf16* tile = &smem[cur][0];
#pragma unroll
    for (int sub = 0; sub < 2; ++sub) {
      const __bf16* pb = tile + (sub * 16 + lm) * D_ + hf * 8;
      v8f ca = {}, cbv = {};  // two chains for WMMA ILP
#pragma unroll
      for (int s = 0; s < 16; s += 2) {
        v16bf bA = load_frag(pb + s * 32);
        v16bf bB = load_frag(pb + (s + 1) * 32);
        ca  = __builtin_amdgcn_wmma_f32_16x16x32_bf16(false, a[s],     false, bA, (short)0, ca,  false, false);
        cbv = __builtin_amdgcn_wmma_f32_16x16x32_bf16(false, a[s + 1], false, bB, (short)0, cbv, false, false);
      }
      const int idx = n + sub * 16 + lm;
      const float cn = cnorm[idx];
#pragma unroll
      for (int j = 0; j < 8; ++j) {
        float score = cn - 2.0f * (ca[j] + cbv[j]);
        if (score < best[j]) { best[j] = score; bidx[j] = idx; }
      }
    }

    if (wave == 0) __builtin_amdgcn_s_wait_tensorcnt(0);
    __syncthreads();  // next tile visible; current buffer free for reuse
    cur ^= 1;
  }

  // Cross-lane argmin within each 16-lane half (all cols of a row live there).
#pragma unroll
  for (int off = 8; off > 0; off >>= 1) {
#pragma unroll
    for (int j = 0; j < 8; ++j) {
      float ov = __shfl_xor(best[j], off, 16);
      int   oi = __shfl_xor(bidx[j], off, 16);
      if (ov < best[j] || (ov == best[j] && oi < bidx[j])) { best[j] = ov; bidx[j] = oi; }
    }
  }
  if (lm == 0) {
#pragma unroll
    for (int j = 0; j < 8; ++j) {
      const int row = mBase + 8 * hf + j;
      bestVal[(size_t)row * NSPLIT + blockIdx.y] = best[j];
      bestIdx[(size_t)row * NSPLIT + blockIdx.y] = bidx[j];
    }
  }
}

// ---------------- kernel 3: combine splits, gather, loss, outputs ----------------
__global__ __launch_bounds__(256) void gather_out_kernel(const float* __restrict__ ze,
                                                         const float* __restrict__ codebook,
                                                         const float* __restrict__ bestVal,
                                                         const int* __restrict__ bestIdx,
                                                         float* __restrict__ out) {
  __shared__ float red[256];
  const int b = blockIdx.x;
  const int t = threadIdx.x;

  float bv = bestVal[(size_t)b * NSPLIT];
  int   bi = bestIdx[(size_t)b * NSPLIT];
#pragma unroll
  for (int s = 1; s < NSPLIT; ++s) {
    float v = bestVal[(size_t)b * NSPLIT + s];
    int   i = bestIdx[(size_t)b * NSPLIT + s];
    if (v < bv || (v == bv && i < bi)) { bv = v; bi = i; }
  }

  const float* crow = codebook + (size_t)bi * D_;
  const float* zrow = ze + (size_t)b * D_;
  float* outz = out + (size_t)b * D_;
  float acc = 0.0f;
#pragma unroll
  for (int d = t; d < D_; d += 256) {
    float zq = crow[d];
    float zev = zrow[d];
    float df = zq - zev;
    outz[d] = zev + df;  // straight-through: z_e + (z_q - z_e)
    acc += df * df;
  }
  red[t] = acc;
  __syncthreads();
#pragma unroll
  for (int s = 128; s > 0; s >>= 1) {
    if (t < s) red[t] += red[t + s];
    __syncthreads();
  }
  if (t == 0) {
    out[(size_t)B_ * D_ + b] = __int_as_float(bi);           // indices (int32 bits)
    out[(size_t)B_ * D_ + B_ + b] = red[0] * (1.0f / D_);    // vq_loss
  }
}

// ---------------- workspace layout ----------------
static constexpr size_t OFF_XB  = 0;
static constexpr size_t SZ_XB   = (size_t)B_ * DIN_ * 2;
static constexpr size_t OFF_WB  = OFF_XB + SZ_XB;
static constexpr size_t SZ_WB   = (size_t)D_ * DIN_ * 2;
static constexpr size_t OFF_CBB = OFF_WB + SZ_WB;
static constexpr size_t SZ_CBB  = (size_t)K_ * D_ * 2;
static constexpr size_t OFF_ZE  = OFF_CBB + SZ_CBB;
static constexpr size_t SZ_ZE   = (size_t)B_ * D_ * 4;
static constexpr size_t OFF_CN  = OFF_ZE + SZ_ZE;
static constexpr size_t SZ_CN   = (size_t)K_ * 4;
static constexpr size_t OFF_BV  = OFF_CN + SZ_CN;
static constexpr size_t SZ_BV   = (size_t)B_ * NSPLIT * 4;
static constexpr size_t OFF_BI  = OFF_BV + SZ_BV;

extern "C" void kernel_launch(void* const* d_in, const int* in_sizes, int n_in,
                              void* d_out, int out_size, void* d_ws, size_t ws_size,
                              hipStream_t stream) {
  (void)in_sizes; (void)n_in; (void)out_size; (void)ws_size;
  const float* x        = (const float*)d_in[0];
  const float* W        = (const float*)d_in[1];
  const float* codebook = (const float*)d_in[2];
  float* out = (float*)d_out;
  char* ws = (char*)d_ws;

  __bf16* xb    = (__bf16*)(ws + OFF_XB);
  __bf16* wb    = (__bf16*)(ws + OFF_WB);
  __bf16* cbb   = (__bf16*)(ws + OFF_CBB);
  float*  ze    = (float*)(ws + OFF_ZE);
  float*  cnorm = (float*)(ws + OFF_CN);
  float*  bv    = (float*)(ws + OFF_BV);
  int*    bi    = (int*)(ws + OFF_BI);

  // 0) precision conversion + codebook norms
  {
    int n4 = (B_ * DIN_) / 4;
    cvt_bf16_kernel<<<n4 / 256, 256, 0, stream>>>(x, xb, n4);
  }
  {
    int n4 = (D_ * DIN_) / 4;
    cvt_bf16_kernel<<<n4 / 256, 256, 0, stream>>>(W, wb, n4);
  }
  cb_prep_kernel<<<K_, 128, 0, stream>>>(codebook, cbb, cnorm);

  // 1) z_e = x @ W^T
  gemm1_kernel<<<dim3(B_ / 128, D_ / 64), 256, 0, stream>>>(xb, wb, ze);

  // 2) distances + argmin (TDM-staged codebook, 4-way split)
  gemm2_argmin_kernel<<<dim3(B_ / 128, NSPLIT), 256, 0, stream>>>(ze, cbb, cnorm, bv, bi);

  // 3) combine, gather, loss, write outputs
  gather_out_kernel<<<B_, 256, 0, stream>>>(ze, codebook, bv, bi, out);
}
